// WeightOnlyQuipLinear_2250562863405
// MI455X (gfx1250) — matrix operand order repaired and checked by reading, not compiled, not run on hardware
//
#include <hip/hip_runtime.h>
#include <hip/hip_bf16.h>
#include <stdint.h>

typedef __attribute__((ext_vector_type(16))) _Float16 v16h;
typedef __attribute__((ext_vector_type(8)))  _Float16 v8h;
typedef __attribute__((ext_vector_type(8)))  float    v8f;

#define IN_FEAT   4096
#define OUT_FEAT  8192
#define TOKENS    64
#define QCOLS     (IN_FEAT / 8)   // 512 codes per output row

// ---------------------------------------------------------------------------
// Kernel 1: x = FWHT4096(input * SU) * Wscale / 64  -> fp16
// One block per token row. 256 threads, 16 KB LDS.
// ---------------------------------------------------------------------------
__global__ void __launch_bounds__(256)
quip_fwht_in_kernel(const float* __restrict__ input,
                    const float* __restrict__ SU,
                    const float* __restrict__ Wscale,
                    _Float16* __restrict__ xh)
{
    __shared__ float s[IN_FEAT];
    const int row = blockIdx.x;
    const int tid = threadIdx.x;
    const float* in = input + row * IN_FEAT;

    for (int i = tid; i < IN_FEAT; i += 256)
        s[i] = in[i] * SU[i];

    for (int h = 1; h < IN_FEAT; h <<= 1) {
        __syncthreads();
        for (int p = tid; p < IN_FEAT / 2; p += 256) {
            int i = ((p & ~(h - 1)) << 1) | (p & (h - 1));
            float a = s[i];
            float b = s[i + h];
            s[i]     = a + b;
            s[i + h] = a - b;
        }
    }
    __syncthreads();

    const float sc = Wscale[0] * (1.0f / 64.0f);   // 1/sqrt(4096) folded in
    _Float16* xr = xh + row * IN_FEAT;
    for (int i = tid; i < IN_FEAT; i += 256)
        xr[i] = (_Float16)(s[i] * sc);
}

// ---------------------------------------------------------------------------
// E8P decode: one 16-bit code -> 8 fp16 weights.
// tbl[idx] packs grid_abs bytes b0..b3 in .x, b4..b7 in .y (low bits),
// parity bit ((sum b)>>1 & 1) stashed at .y bit 31.
// ---------------------------------------------------------------------------
__device__ __forceinline__ void e8p_decode8(unsigned code, const uint2* __restrict__ tbl,
                                            _Float16* __restrict__ out)
{
    code &= 0xFFFFu;
    const unsigned abs_idx = code & 255u;
    const unsigned sbits   = (code >> 8) & 127u;
    const unsigned shift   = (code >> 15) & 1u;
    const uint2    p       = tbl[abs_idx];
    const unsigned parity  = p.y >> 31;
    const unsigned w0      = p.x;
    const unsigned w1      = p.y & 0x7FFFFFFFu;
    const unsigned s8      = (__popc(sbits) ^ parity) & 1u;
    const unsigned signs   = sbits | (s8 << 7);
    const float    shiftv  = 0.25f - 0.5f * (float)shift;
#pragma unroll
    for (int i = 0; i < 8; ++i) {
        unsigned b = ((i < 4) ? (w0 >> (8 * i)) : (w1 >> (8 * (i - 4)))) & 255u;
        float w = 0.5f * (float)b;                 // odd b -> 0.5 .. 3.5
        w = ((signs >> i) & 1u) ? -w : w;
        out[i] = (_Float16)(w + shiftv);
    }
}

// ---------------------------------------------------------------------------
// Kernel 2: out(64 x 8192) = x(64x4096 f16) @ decode(Qidxs)^T, fp32 accum.
// 128 blocks x 128 threads (4 waves). Each wave: one N-tile of 16 output
// features x all 64 tokens (4 wmma accumulators), K swept in steps of 32.
// v_wmma_f32_16x16x32_f16, each E8P code decoded exactly once.
// ---------------------------------------------------------------------------
__global__ void __launch_bounds__(128)
quip_e8p_gemm_kernel(const _Float16* __restrict__ xh,
                     const int*      __restrict__ Qidxs,
                     const int*      __restrict__ grid_abs,
                     float*          __restrict__ gout)
{
    __shared__ uint2 tbl[256];

    const int tid = threadIdx.x;
    // Build decode table: pack 8 abs-bytes + parity bit per grid entry.
    for (int e = tid; e < 256; e += 128) {
        const int* g = grid_abs + e * 8;
        unsigned lo = 0, hi = 0, sum = 0;
#pragma unroll
        for (int j = 0; j < 4; ++j) {
            unsigned bl = (unsigned)g[j];
            unsigned bh = (unsigned)g[4 + j];
            lo |= bl << (8 * j);
            hi |= bh << (8 * j);
            sum += bl + bh;
        }
        unsigned parity = (sum >> 1) & 1u;
        tbl[e] = make_uint2(lo, hi | (parity << 31));
    }
    __syncthreads();

    const int wave  = tid >> 5;
    const int lane  = tid & 31;
    const int l16   = lane & 15;
    const int khalf = lane >> 4;          // which half-wave (K split)
    const int n0    = (blockIdx.x * 4 + wave) * 16;
    const int n     = n0 + l16;           // this lane's output feature

    const int* __restrict__ qrow = Qidxs + (size_t)n * QCOLS;

    v8f acc0 = {}, acc1 = {}, acc2 = {}, acc3 = {};

    union AV { v16h v; v8h h[2]; };
    union BV { v16h v; _Float16 e[16]; };

    for (int k0 = 0; k0 < IN_FEAT; k0 += 32) {
        // ---- A fragments: 16x32, contiguous K order shared with B.
        // lanes 0-15: K = k0..k0+15 ; lanes 16-31: K = k0+16..k0+31.
        const int kb = k0 + khalf * 16;
        AV a0, a1, a2, a3;
        const _Float16* xa = xh + (size_t)l16 * IN_FEAT + kb;
        a0.h[0] = *(const v8h*)(xa + 0 * 16 * IN_FEAT);
        a0.h[1] = *(const v8h*)(xa + 0 * 16 * IN_FEAT + 8);
        a1.h[0] = *(const v8h*)(xa + 1 * 16 * IN_FEAT);
        a1.h[1] = *(const v8h*)(xa + 1 * 16 * IN_FEAT + 8);
        a2.h[0] = *(const v8h*)(xa + 2 * 16 * IN_FEAT);
        a2.h[1] = *(const v8h*)(xa + 2 * 16 * IN_FEAT + 8);
        a3.h[0] = *(const v8h*)(xa + 3 * 16 * IN_FEAT);
        a3.h[1] = *(const v8h*)(xa + 3 * 16 * IN_FEAT + 8);

        // ---- B fragment: 32x16, lanes 0-15 hold K=k0..k0+15 for col N=lane,
        // lanes 16-31 hold K=k0+16..k0+31. Two codes per lane.
        const int j = (k0 + khalf * 16) >> 3;        // code index (multiple of 2)
        const uint2 q = *(const uint2*)(qrow + j);
        __builtin_prefetch(qrow + j + 4, 0, 0);      // next K-step's codes

        BV b;
        e8p_decode8((unsigned)q.x, tbl, &b.e[0]);
        e8p_decode8((unsigned)q.y, tbl, &b.e[8]);

        acc0 = __builtin_amdgcn_wmma_f32_16x16x32_f16(false, a0.v, false, b.v,
                                                      (short)0, acc0, false, false);
        acc1 = __builtin_amdgcn_wmma_f32_16x16x32_f16(false, a1.v, false, b.v,
                                                      (short)0, acc1, false, false);
        acc2 = __builtin_amdgcn_wmma_f32_16x16x32_f16(false, a2.v, false, b.v,
                                                      (short)0, acc2, false, false);
        acc3 = __builtin_amdgcn_wmma_f32_16x16x32_f16(false, a3.v, false, b.v,
                                                      (short)0, acc3, false, false);
    }

    // ---- Store: 32-bit C/D 16x16 layout: lane<16 -> M=v, lane>=16 -> M=v+8,
    // N = lane & 15.
#pragma unroll
    for (int v = 0; v < 8; ++v) {
        const int m = v + khalf * 8;
        gout[(size_t)(0 * 16 + m) * OUT_FEAT + n] = acc0[v];
        gout[(size_t)(1 * 16 + m) * OUT_FEAT + n] = acc1[v];
        gout[(size_t)(2 * 16 + m) * OUT_FEAT + n] = acc2[v];
        gout[(size_t)(3 * 16 + m) * OUT_FEAT + n] = acc3[v];
    }
}

// ---------------------------------------------------------------------------
// Kernel 3: out = FWHT8192(gemm_out) / sqrt(8192) * SV + bias
// One block per token row. 256 threads, 32 KB LDS.
// ---------------------------------------------------------------------------
__global__ void __launch_bounds__(256)
quip_fwht_out_kernel(const float* __restrict__ gout,
                     const float* __restrict__ SV,
                     const float* __restrict__ bias,
                     float* __restrict__ out)
{
    __shared__ float s[OUT_FEAT];
    const int row = blockIdx.x;
    const int tid = threadIdx.x;
    const float* gr = gout + (size_t)row * OUT_FEAT;

    for (int i = tid; i < OUT_FEAT; i += 256)
        s[i] = gr[i];

    for (int h = 1; h < OUT_FEAT; h <<= 1) {
        __syncthreads();
        for (int p = tid; p < OUT_FEAT / 2; p += 256) {
            int i = ((p & ~(h - 1)) << 1) | (p & (h - 1));
            float a = s[i];
            float b = s[i + h];
            s[i]     = a + b;
            s[i + h] = a - b;
        }
    }
    __syncthreads();

    const float sc = 0.011048543456039806f;  // 1/sqrt(8192)
    float* orow = out + (size_t)row * OUT_FEAT;
    for (int i = tid; i < OUT_FEAT; i += 256)
        orow[i] = s[i] * sc * SV[i] + bias[i];
}

// ---------------------------------------------------------------------------
extern "C" void kernel_launch(void* const* d_in, const int* in_sizes, int n_in,
                              void* d_out, int out_size, void* d_ws, size_t ws_size,
                              hipStream_t stream)
{
    (void)in_sizes; (void)n_in; (void)out_size; (void)ws_size;

    const float* input    = (const float*)d_in[0];   // (64, 4096) f32
    const int*   Qidxs    = (const int*)  d_in[1];   // (8192, 512) i32
    const int*   grid_abs = (const int*)  d_in[2];   // (256, 8) i32
    const float* SU       = (const float*)d_in[3];   // (4096,) f32
    const float* SV       = (const float*)d_in[4];   // (8192,) f32
    const float* Wscale   = (const float*)d_in[5];   // scalar f32
    const float* bias     = (const float*)d_in[6];   // (8192,) f32
    float*       out      = (float*)d_out;           // (64, 8192) f32

    // Workspace layout: fp16 x (512 KB) | fp32 gemm output (2 MB)
    _Float16* xh   = (_Float16*)d_ws;
    float*    gout = (float*)((char*)d_ws + (size_t)TOKENS * IN_FEAT * sizeof(_Float16));

    quip_fwht_in_kernel<<<TOKENS, 256, 0, stream>>>(input, SU, Wscale, xh);

    // 512 N-tiles of 16 features; 4 waves/block -> 128 blocks.
    quip_e8p_gemm_kernel<<<OUT_FEAT / 64, 128, 0, stream>>>(xh, Qidxs, grid_abs, gout);

    quip_fwht_out_kernel<<<TOKENS, 256, 0, stream>>>(gout, SV, bias, out);
}
